// ResMemNet_7275674600021
// MI455X (gfx1250) — matrix operand order repaired and checked by reading, not compiled, not run on hardware
//
#include <hip/hip_runtime.h>
#include <hip/hip_bf16.h>

// ---------------------------------------------------------------------------
// ResMemNet for MI455X (gfx1250), wave32 + WMMA bf16 + async global->LDS.
//
// Exact algebraic simplifications (identities, not approximations):
//   * sum(softmax(s), axis=-1) == 1  -> attention head contributes s==1.
//   * mem @ Wout.T with mem = s (x) vmean  ->  constant row vector
//       r[d] = sum_j vmean[j%256] * Wout[d, j]      (per layer)
//   So per block:  h = LayerNorm( relu(h @ W.T + b) + r ).
// ---------------------------------------------------------------------------

typedef unsigned short u16;
typedef __attribute__((ext_vector_type(8)))  u16    u16x8;
typedef __attribute__((ext_vector_type(16))) __bf16 bf16x16;
typedef __attribute__((ext_vector_type(8)))  float  v8f;
typedef int v4i_g __attribute__((vector_size(16)));   // matches builtin param type

#define B_ROWS 16384
#define D_DIM  1024
#define L_NUM  3
#define NKV    128
#define V_DIM  256
#define OUT_N  256
#define KF     (L_NUM * D_DIM)   // 3072

#ifndef __has_builtin
#define __has_builtin(x) 0
#endif
#if __has_builtin(__builtin_amdgcn_global_load_async_to_lds_b128)
#define ASYNC_LDS 1
#else
#define ASYNC_LDS 0
#endif
#if __has_builtin(__builtin_amdgcn_s_wait_asynccnt)
#define HAVE_WAIT_ASYNC 1
#else
#define HAVE_WAIT_ASYNC 0
#endif

__device__ __forceinline__ u16 f32_to_bf16(float f) {
  unsigned u = __builtin_bit_cast(unsigned, f);
  unsigned r = u + 0x7FFFu + ((u >> 16) & 1u);   // round-to-nearest-even
  return (u16)(r >> 16);
}
__device__ __forceinline__ float bf16_to_f32(u16 h) {
  return __builtin_bit_cast(float, ((unsigned)h) << 16);
}

#if ASYNC_LDS
// 16B global -> LDS async copy (GLOBAL_LOAD_ASYNC_TO_LDS_B128, ASYNCcnt).
__device__ __forceinline__ void async_cp16(const u16* g, u16* l) {
  __builtin_amdgcn_global_load_async_to_lds_b128(
      (__attribute__((address_space(1))) v4i_g*)g,
      (__attribute__((address_space(3))) v4i_g*)l,
      0, 0);
}
#endif

template <int N>
__device__ __forceinline__ void wait_async() {
#if ASYNC_LDS
#if HAVE_WAIT_ASYNC
  __builtin_amdgcn_s_wait_asynccnt(N);
#else
  asm volatile("s_wait_asynccnt %0" ::"i"(N) : "memory");
#endif
#endif
}

struct Pack32 { u16x8 a, b; };

// A fragment, 16-bit A 16x32 (ISA 7.12.2): lane(l&15)=row M,
// lanes<16 hold K {0..7, 16..23}, lanes>=16 hold K {8..15, 24..31}.
__device__ __forceinline__ bf16x16 load_fragA(const u16* rowbase, int lanehi) {
  const int kb = lanehi * 8;
  u16x8 lo = *(const u16x8*)(rowbase + kb);
  u16x8 hi = *(const u16x8*)(rowbase + kb + 16);
  Pack32 p{lo, hi};
  return __builtin_bit_cast(bf16x16, p);
}

// B fragment, 16-bit B 32x16 (ISA 7.12.4 pattern): lane(l&15)=col N,
// lanes<16 hold K {0..7, 16..23}, lanes>=16 hold K {8..15, 24..31}.
__device__ __forceinline__ bf16x16 load_fragB(const u16* rowbase, int lanehi) {
  const int kb = lanehi * 8;
  u16x8 lo = *(const u16x8*)(rowbase + kb);
  u16x8 hi = *(const u16x8*)(rowbase + kb + 16);
  Pack32 p{lo, hi};
  return __builtin_bit_cast(bf16x16, p);
}

// --------------------------- f32 -> bf16 convert ---------------------------
__global__ void k_cvt_bf16(const float* __restrict__ in, u16* __restrict__ out, int n) {
  int i = blockIdx.x * blockDim.x + threadIdx.x;
  if (i < n) out[i] = f32_to_bf16(in[i]);
}

// --------------------- vmean[l][v] = mean_n values[l,n,v] ------------------
__global__ __launch_bounds__(256) void k_vmean(const float* __restrict__ values,
                                               float* __restrict__ vmean) {
  int l = blockIdx.x;
  int v = threadIdx.x;
  const float* p = values + (size_t)l * NKV * V_DIM + v;
  float s = 0.f;
  for (int n = 0; n < NKV; ++n) s += p[(size_t)n * V_DIM];
  vmean[l * V_DIM + v] = s * (1.0f / NKV);
}

// ---------------- r[l][d] = sum_j vmean[l][j&255] * Wout[l,d,j] ------------
__global__ __launch_bounds__(256) void k_rvec(const float* __restrict__ Wout,
                                              const float* __restrict__ vmean,
                                              float* __restrict__ rv) {
  __shared__ float vm[V_DIM];
  int idx = blockIdx.x * 256 + threadIdx.x;      // block stays within one l
  int l = idx >> 10;
  int d = idx & (D_DIM - 1);
  vm[threadIdx.x] = vmean[l * V_DIM + threadIdx.x];
  __syncthreads();
  const float* w = Wout + ((size_t)l * D_DIM + d) * (size_t)D_DIM;
  float s = 0.f;
  for (int j = 0; j < D_DIM; ++j) s += vm[j & (V_DIM - 1)] * w[j];
  rv[idx] = s;
}

// ------------------------- bf16 WMMA GEMM ----------------------------------
// C(row,col) = act( A[row,:] . Wt[col,:] + bias[col] )
// A: (M x K) bf16 row-major (lda). Wt: (N x K) bf16 row-major.
// Block tile (WAVES_M*MT*16) x (WAVES_N*NT*16), 8 waves, wave tile (MT*16)x(NT*16).
// KTILE=32, double-buffered LDS fed by async global->LDS copies.
template <int WAVES_M, int WAVES_N, int MT, int NT>
__global__ __launch_bounds__(256)
void k_gemm_bf16(const u16* __restrict__ A, int lda,
                 const u16* __restrict__ Wt,
                 const float* __restrict__ bias,
                 float* __restrict__ outf, int ldf,
                 u16* __restrict__ outb, int ldob,
                 int K, int do_relu) {
  static_assert(WAVES_M * WAVES_N == 8, "8 waves per block");
  constexpr int BM  = WAVES_M * MT * 16;          // block rows
  constexpr int BN  = WAVES_N * NT * 16;          // block cols
  constexpr int LDT = 40;                         // u16 per LDS row (80B, 16B-aligned)
  constexpr int CA  = BM / 64;                    // A copy chunks per thread/stage
  constexpr int CB  = BN / 64;                    // B copy chunks per thread/stage
  __shared__ __align__(16) u16 At[2][BM * LDT];
  __shared__ __align__(16) u16 Bt[2][BN * LDT];

  const int tid    = threadIdx.x;
  const int lane   = tid & 31;
  const int wid    = tid >> 5;
  const int wm     = wid % WAVES_M;
  const int wn     = wid / WAVES_M;
  const int lanelo = lane & 15;
  const int lanehi = lane >> 4;
  const int rowBase = blockIdx.y * BM;
  const int colBase = blockIdx.x * BN;

  auto issue = [&](int k0, int buf) {
#pragma unroll
    for (int t = 0; t < CA; ++t) {
      int c = tid + t * 256, r = c >> 2, c4 = c & 3;
      const u16* g = A + (size_t)(rowBase + r) * lda + k0 + c4 * 8;
      u16*       l = &At[buf][r * LDT + c4 * 8];
#if ASYNC_LDS
      async_cp16(g, l);
#else
      *(u16x8*)l = *(const u16x8*)g;
#endif
    }
#pragma unroll
    for (int t = 0; t < CB; ++t) {
      int c = tid + t * 256, r = c >> 2, c4 = c & 3;
      const u16* g = Wt + (size_t)(colBase + r) * K + k0 + c4 * 8;
      u16*       l = &Bt[buf][r * LDT + c4 * 8];
#if ASYNC_LDS
      async_cp16(g, l);
#else
      *(u16x8*)l = *(const u16x8*)g;
#endif
    }
  };

  v8f acc[MT][NT];
  v8f vzero = {0.f, 0.f, 0.f, 0.f, 0.f, 0.f, 0.f, 0.f};
#pragma unroll
  for (int m = 0; m < MT; ++m)
#pragma unroll
    for (int n = 0; n < NT; ++n) acc[m][n] = vzero;

  const int nstage = K / 32;
  issue(0, 0);
  for (int s = 0; s < nstage; ++s) {
    const int cur = s & 1;
    const bool more = (s + 1 < nstage);
    if (more) issue((s + 1) * 32, cur ^ 1);       // next stage in flight
    if (more) wait_async<CA + CB>(); else wait_async<0>();
    __syncthreads();                              // stage s tiles visible to all

    bf16x16 fa[MT], fb[NT];
#pragma unroll
    for (int m = 0; m < MT; ++m)
      fa[m] = load_fragA(&At[cur][(wm * MT * 16 + m * 16 + lanelo) * LDT], lanehi);
#pragma unroll
    for (int n = 0; n < NT; ++n)
      fb[n] = load_fragB(&Bt[cur][(wn * NT * 16 + n * 16 + lanelo) * LDT], lanehi);
#pragma unroll
    for (int m = 0; m < MT; ++m)
#pragma unroll
      for (int n = 0; n < NT; ++n)
        acc[m][n] = __builtin_amdgcn_wmma_f32_16x16x32_bf16(
            false, fa[m], false, fb[n], (short)0, acc[m][n], false, false);

    __syncthreads();                              // done reading buf before reuse
  }

  // Epilogue. C/D layout: lanes 0-15: N=lane, M=r; lanes 16-31: N=lane-16, M=8+r.
#pragma unroll
  for (int m = 0; m < MT; ++m) {
#pragma unroll
    for (int n = 0; n < NT; ++n) {
      int col  = colBase + wn * NT * 16 + n * 16 + lanelo;
      int row0 = rowBase + wm * MT * 16 + m * 16 + lanehi * 8;
      float bv = bias ? bias[col] : 0.f;
#pragma unroll
      for (int r = 0; r < 8; ++r) {
        float v = acc[m][n][r] + bv;
        if (do_relu) v = v > 0.f ? v : 0.f;
        int row = row0 + r;
        if (outf) outf[(size_t)row * ldf  + col] = v;
        if (outb) outb[(size_t)row * ldob + col] = f32_to_bf16(v);
      }
    }
  }
}

// ---------- h = LayerNorm(bf16 h1 + r) -> bf16 into concat buffer ----------
__global__ __launch_bounds__(256)
void k_add_ln(const u16* __restrict__ h1, const float* __restrict__ r,
              const float* __restrict__ g, const float* __restrict__ be,
              u16* __restrict__ outb, int ldz) {
  __shared__ float rs[256], rq[256];
  int row = blockIdx.x;
  const u16* hr = h1 + (size_t)row * D_DIM;
  float x[4], s = 0.f, sq = 0.f;
#pragma unroll
  for (int j = 0; j < 4; ++j) {
    int d = threadIdx.x + j * 256;
    float v = bf16_to_f32(hr[d]) + r[d];
    x[j] = v; s += v; sq += v * v;
  }
  rs[threadIdx.x] = s; rq[threadIdx.x] = sq;
  __syncthreads();
  for (int off = 128; off > 0; off >>= 1) {
    if (threadIdx.x < off) {
      rs[threadIdx.x] += rs[threadIdx.x + off];
      rq[threadIdx.x] += rq[threadIdx.x + off];
    }
    __syncthreads();
  }
  float mean = rs[0] * (1.0f / D_DIM);
  float var  = rq[0] * (1.0f / D_DIM) - mean * mean;
  float inv  = rsqrtf(var + 1e-5f);
  u16* orow = outb + (size_t)row * ldz;
#pragma unroll
  for (int j = 0; j < 4; ++j) {
    int d = threadIdx.x + j * 256;
    orow[d] = f32_to_bf16((x[j] - mean) * inv * g[d] + be[d]);
  }
}

// ------------------- final LayerNorm + sigmoid -> f32 out ------------------
__global__ __launch_bounds__(256)
void k_lnsig(const float* __restrict__ z, const float* __restrict__ g,
             const float* __restrict__ be, float* __restrict__ out) {
  __shared__ float rs[256], rq[256];
  int row = blockIdx.x;
  float v = z[(size_t)row * OUT_N + threadIdx.x];
  rs[threadIdx.x] = v; rq[threadIdx.x] = v * v;
  __syncthreads();
  for (int off = 128; off > 0; off >>= 1) {
    if (threadIdx.x < off) {
      rs[threadIdx.x] += rs[threadIdx.x + off];
      rq[threadIdx.x] += rq[threadIdx.x + off];
    }
    __syncthreads();
  }
  float mean = rs[0] * (1.0f / OUT_N);
  float var  = rq[0] * (1.0f / OUT_N) - mean * mean;
  float y = (v - mean) * rsqrtf(var + 1e-5f) * g[threadIdx.x] + be[threadIdx.x];
  out[(size_t)row * OUT_N + threadIdx.x] = 1.0f / (1.0f + __expf(-y));
}

// ---------------------------------------------------------------------------
extern "C" void kernel_launch(void* const* d_in, const int* in_sizes, int n_in,
                              void* d_out, int out_size, void* d_ws, size_t ws_size,
                              hipStream_t stream) {
  const float* x      = (const float*)d_in[0];
  const float* W      = (const float*)d_in[1];
  const float* bb     = (const float*)d_in[2];
  // d_in[3] Wq, d_in[4] keys: unused (softmax-row-sum identity == 1)
  const float* values = (const float*)d_in[5];
  const float* Wout   = (const float*)d_in[6];
  const float* lng    = (const float*)d_in[7];
  const float* lnb    = (const float*)d_in[8];
  const float* Wf     = (const float*)d_in[9];
  const float* bf     = (const float*)d_in[10];
  const float* lnfg   = (const float*)d_in[11];
  const float* lnfb   = (const float*)d_in[12];

  auto al = [](size_t v) { return (v + 255) & ~(size_t)255; };
  char* p = (char*)d_ws;
  size_t off = 0;
  u16*   xb  = (u16*)  (p + off); off = al(off + (size_t)B_ROWS * D_DIM * 2);
  u16*   Wb  = (u16*)  (p + off); off = al(off + (size_t)L_NUM * D_DIM * D_DIM * 2);
  u16*   Wfb = (u16*)  (p + off); off = al(off + (size_t)OUT_N * KF * 2);
  u16*   h1b = (u16*)  (p + off); off = al(off + (size_t)B_ROWS * D_DIM * 2);
  u16*   zb  = (u16*)  (p + off); off = al(off + (size_t)B_ROWS * KF * 2);
  float* zf  = (float*)(p + off); off = al(off + (size_t)B_ROWS * OUT_N * 4);
  float* vm  = (float*)(p + off); off = al(off + (size_t)L_NUM * V_DIM * 4);
  float* rv  = (float*)(p + off); off = al(off + (size_t)L_NUM * D_DIM * 4);
  (void)ws_size; (void)in_sizes; (void)n_in; (void)out_size;

  // 1) convert inputs/weights to bf16
  {
    int n0 = B_ROWS * D_DIM;
    k_cvt_bf16<<<(n0 + 255) / 256, 256, 0, stream>>>(x, xb, n0);
    int n1 = L_NUM * D_DIM * D_DIM;
    k_cvt_bf16<<<(n1 + 255) / 256, 256, 0, stream>>>(W, Wb, n1);
    int n2 = OUT_N * KF;
    k_cvt_bf16<<<(n2 + 255) / 256, 256, 0, stream>>>(Wf, Wfb, n2);
  }

  // 2) collapsed MemLayer constants
  k_vmean<<<L_NUM, 256, 0, stream>>>(values, vm);
  k_rvec <<<(L_NUM * D_DIM) / 256, 256, 0, stream>>>(Wout, vm, rv);

  // 3) three residual blocks: GEMM (128x256 blocks, 64x64 wave tiles) + LN
  for (int i = 0; i < L_NUM; ++i) {
    const u16* Ain = (i == 0) ? xb : (zb + (size_t)(i - 1) * D_DIM);
    int lda        = (i == 0) ? D_DIM : KF;
    k_gemm_bf16<2, 4, 4, 4><<<dim3(D_DIM / 256, B_ROWS / 128), 256, 0, stream>>>(
        Ain, lda, Wb + (size_t)i * D_DIM * D_DIM, bb + (size_t)i * D_DIM,
        nullptr, 0, h1b, D_DIM, D_DIM, /*relu=*/1);
    k_add_ln<<<B_ROWS, 256, 0, stream>>>(
        h1b, rv + (size_t)i * D_DIM, lng + (size_t)i * D_DIM, lnb + (size_t)i * D_DIM,
        zb + (size_t)i * D_DIM, KF);
  }

  // 4) head GEMM (128x128 blocks keeps 256 workgroups), then LN + sigmoid
  k_gemm_bf16<4, 2, 2, 4><<<dim3(OUT_N / 128, B_ROWS / 128), 256, 0, stream>>>(
      zb, KF, Wfb, bf, zf, OUT_N, nullptr, 0, KF, /*relu=*/0);
  k_lnsig<<<B_ROWS, 256, 0, stream>>>(zf, lnfg, lnfb, (float*)d_out);
}
